// Quantizer_41781441855853
// MI455X (gfx1250) — matrix operand order repaired and checked by reading, not compiled, not run on hardware
//
#include <hip/hip_runtime.h>
#include <stdint.h>

// ---------------- CDNA5 WMMA types ----------------
typedef __attribute__((ext_vector_type(16))) __bf16 v16bf;
typedef __attribute__((ext_vector_type(8)))  float  v8f;

#define C_DIM      256     // embedding dim
#define HW         1024    // 32*32 pixels per image
#define TILE_CODES 16      // codes per WMMA tile (M of D)
#define KCHUNK     32      // K per v_wmma_f32_16x16x32_bf16
#define NCHUNKS    (C_DIM / KCHUNK)   // 8
#define WG_PIXELS  128     // pixels per workgroup (8 waves x 16)

__device__ __forceinline__ unsigned pack2_bf16(float a, float b) {
  // round-to-nearest-even fp32 -> bf16, packed pair
  unsigned ua = __float_as_uint(a);
  unsigned ub = __float_as_uint(b);
  ua = (ua + 0x7FFFu + ((ua >> 16) & 1u)) >> 16;
  ub = (ub + 0x7FFFu + ((ub >> 16) & 1u)) >> 16;
  return (ua & 0xFFFFu) | (ub << 16);
}

// -------------------------------------------------------------------------
// Prep 1: e[K,256] fp32 -> bf16 fragments, pre-swizzled to the 16-bit
// A-matrix 16x32 per-lane layout (ISA 7.12.2):
//   lanes 0-15 : row M=lane,   K = {0..7, 16..23} (+half*8 for lanes 16-31)
// One thread per (tile, kchunk, lane); writes its 32B lane slot.
// -------------------------------------------------------------------------
__global__ void vq_prep_efrag(const float* __restrict__ e,
                              uint4* __restrict__ efrag_q4, int K) {
  int g    = blockIdx.x * blockDim.x + threadIdx.x;
  int lane = g & 31;
  int q    = (g >> 5) & (NCHUNKS - 1);
  int tile = g >> 8;
  int half = lane >> 4;
  int code = tile * TILE_CODES + (lane & 15);
  if (code >= K) return;
  const float* er = e + (size_t)code * C_DIM + q * KCHUNK;
  unsigned w[8];
#pragma unroll
  for (int tp = 0; tp < 8; ++tp) {
    int j0 = 2 * tp, j1 = 2 * tp + 1;
    int k0 = j0 + ((j0 < 8) ? 0 : 8) + half * 8;
    int k1 = j1 + ((j1 < 8) ? 0 : 8) + half * 8;
    w[tp] = pack2_bf16(er[k0], er[k1]);
  }
  uint4* dst = efrag_q4 + (size_t)g * 2;
  dst[0] = make_uint4(w[0], w[1], w[2], w[3]);
  dst[1] = make_uint4(w[4], w[5], w[6], w[7]);
}

// Prep 2: ee[k] = ||e_k||^2 in exact fp32 (argmin correction term)
__global__ void vq_prep_ee(const float* __restrict__ e,
                           float* __restrict__ ee, int K) {
  int k = blockIdx.x * blockDim.x + threadIdx.x;
  if (k >= K) return;
  const float4* er = (const float4*)(e + (size_t)k * C_DIM);
  float s = 0.f;
#pragma unroll 8
  for (int i = 0; i < C_DIM / 4; ++i) {
    float4 v = er[i];
    s += v.x * v.x + v.y * v.y + v.z * v.z + v.w * v.w;
  }
  ee[k] = s;
}

// -------------------------------------------------------------------------
// Main: D = A(codes 16x32 bf16) x B(pixels 32x16 bf16) + acc, fp32 accum.
// argmin_k( ee[k] - 2*dot )  (||z||^2 dropped: constant per pixel)
// -------------------------------------------------------------------------
union Frag32 {
  unsigned u[8];
  uint4    q4[2];
  v16bf    v;
};

__global__ __launch_bounds__(256) void vq_main(
    const float* __restrict__ x, const float* __restrict__ e,
    const uint4* __restrict__ efrag_q4, const float* __restrict__ ee,
    float* __restrict__ out, int num_tiles) {
  __shared__ uint4    lds_tile[NCHUNKS * 32 * 2];  // 8 KB: one 16-code tile
  __shared__ float    lds_ee[TILE_CODES];
  __shared__ unsigned lds_idx[WG_PIXELS];

  const int tid  = threadIdx.x;
  const int lane = tid & 31;
  const int wave = tid >> 5;
  const int half = lane >> 4;  // 0/1
  const int col  = lane & 15;  // pixel column within tile

  // ---- load this wave's 16-pixel bf16 B-fragments (held in VGPRs) ----
  // B 32x16 layout: lane -> col N = lane&15; VGPRs 0..7 hold K = {0..7, 16..23} + half*8
  long p = (long)blockIdx.x * WG_PIXELS + wave * 16 + col;
  const int b  = (int)(p >> 10);
  const int hw = (int)(p & (HW - 1));
  const float* xb = x + ((long)b << 18) + hw;  // + c*HW per element

  Frag32 zf[NCHUNKS];
#pragma unroll
  for (int q = 0; q < NCHUNKS; ++q) {
#pragma unroll
    for (int tp = 0; tp < 8; ++tp) {
      int c0   = q * KCHUNK + half * 8 + 2 * tp + ((tp < 4) ? 0 : 8);
      float f0 = xb[(size_t)c0 << 10];
      float f1 = xb[(size_t)(c0 + 1) << 10];
      zf[q].u[tp] = pack2_bf16(f0, f1);
    }
  }

  float    best = 3.4e38f;
  unsigned bidx = 0;

  for (int t = 0; t < num_tiles; ++t) {
    __syncthreads();  // previous tile fully consumed
    // cooperative stage: 8 KB tile, 32B per thread
    const uint4* src = efrag_q4 + (size_t)t * (NCHUNKS * 32 * 2);
    lds_tile[tid * 2]     = src[tid * 2];
    lds_tile[tid * 2 + 1] = src[tid * 2 + 1];
    if (tid < TILE_CODES) lds_ee[tid] = ee[t * TILE_CODES + tid];
    __syncthreads();
    if (t + 1 < num_tiles)  // global_prefetch_b8 for next tile
      __builtin_prefetch(efrag_q4 + (size_t)(t + 1) * (NCHUNKS * 32 * 2) + tid * 2, 0, 0);

    v8f acc = {0.f, 0.f, 0.f, 0.f, 0.f, 0.f, 0.f, 0.f};
#pragma unroll
    for (int q = 0; q < NCHUNKS; ++q) {
      Frag32 af;
      af.q4[0] = lds_tile[(q * 32 + lane) * 2];
      af.q4[1] = lds_tile[(q * 32 + lane) * 2 + 1];
      acc = __builtin_amdgcn_wmma_f32_16x16x32_bf16(
          false, af.v, false, zf[q].v, (short)0, acc, false, false);
    }
    // D layout: VGPR r, this lane -> (code m = r + half*8, pixel n = col)
#pragma unroll
    for (int r = 0; r < 8; ++r) {
      float    s  = lds_ee[r + half * 8] - 2.0f * acc[r];
      unsigned id = (unsigned)(t * TILE_CODES + r + half * 8);
      if (s < best) { best = s; bidx = id; }
    }
  }

  // combine half-lanes (codes 0-7 vs 8-15 of each tile) — wave32 shuffle
  float    obest = __shfl_xor(best, 16, 32);
  unsigned oidx  = (unsigned)__shfl_xor((int)bidx, 16, 32);
  if (obest < best || (obest == best && oidx < bidx)) { best = obest; bidx = oidx; }
  if (half == 0) lds_idx[wave * 16 + col] = bidx;
  __syncthreads();

  // ---- gather e[idx] -> out[B,C,H,W], coalesced 128-float rows ----
  {
    int      i   = tid & (WG_PIXELS - 1);
    int      c0  = tid >> 7;  // 0/1: channel parity
    unsigned id  = lds_idx[i];
    long     pp  = (long)blockIdx.x * WG_PIXELS + i;
    int      bb  = (int)(pp >> 10);
    int      hh  = (int)(pp & (HW - 1));
    float*   op  = out + ((long)bb << 18) + hh;
    const float* er = e + (size_t)id * C_DIM;
    for (int c = c0; c < C_DIM; c += 2)
      op[(size_t)c << 10] = er[c];
  }
}

extern "C" void kernel_launch(void* const* d_in, const int* in_sizes, int n_in,
                              void* d_out, int out_size, void* d_ws, size_t ws_size,
                              hipStream_t stream) {
  (void)n_in; (void)out_size; (void)ws_size;
  const float* x = (const float*)d_in[0];
  const float* e = (const float*)d_in[1];
  float*     out = (float*)d_out;

  const int  K         = in_sizes[1] / C_DIM;            // 512
  const long N_pixels  = (long)in_sizes[0] / C_DIM;      // 131072
  const int  num_tiles = K / TILE_CODES;                 // 32

  // workspace: [0, K*C*2)        bf16 pre-swizzled code fragments (256 KB)
  //            [K*C*2, +K*4)     fp32 ||e_k||^2                    (2 KB)
  uint4* efrag = (uint4*)d_ws;
  float* ee    = (float*)((char*)d_ws + (size_t)K * C_DIM * 2);

  const int frag_threads = num_tiles * NCHUNKS * 32;     // 8192
  vq_prep_efrag<<<(frag_threads + 255) / 256, 256, 0, stream>>>(e, efrag, K);
  vq_prep_ee<<<(K + 255) / 256, 256, 0, stream>>>(e, ee, K);

  const int nblocks = (int)(N_pixels / WG_PIXELS);       // 1024
  vq_main<<<nblocks, 256, 0, stream>>>(x, e, efrag, ee, out, num_tiles);
}
